// GlobalNet_v2_69776038690930
// MI455X (gfx1250) — matrix operand order repaired and checked by reading, not compiled, not run on hardware
//
#include <hip/hip_runtime.h>

// ============================================================================
// MI455X (gfx1250) implementation of FPS + sine-MLP + gather.
//
// Key algebraic optimization: take_along_axis(MLP(x), idx) == MLP(gather(x,idx))
// for a per-point MLP, so we run the 6-layer MLP on only the 16*1024 sampled
// points (8x less work than the reference's order, identical semantics).
//
// FPS is the latency-critical part (1024 strictly sequential argmax steps).
// We keep each batch's 8192x6 points AND the running min-distance array fully
// in registers (one 1024-thread block per batch, 8 points/thread), so the
// k-loop touches memory only for the 6-float centroid broadcast (L1-hot).
//
// The MLP is a chain of tiny GEMMs (M=16384, K<=256, N<=512) executed with
// v_wmma_f32_16x16x32_f16 (f32 accumulate), activations ping-ponged in f16
// through workspace. Total traffic ~70MB -> ~3us at 23.3 TB/s; FPS dominates.
// ============================================================================

typedef _Float16 half_t;
typedef __attribute__((ext_vector_type(16))) _Float16 v16h;
typedef __attribute__((ext_vector_type(8)))  float    v8f;

#define B_   16
#define N_   8192
#define D_   6
#define KS   1024
#define M_   (B_ * KS)          // 16384 gathered rows
#define FPS_T 1024
#define PPT  (N_ / FPS_T)       // 8 points per thread

// ---------------------------------------------------------------------------
// Farthest point sampling: one block per batch, points resident in registers.
// ---------------------------------------------------------------------------
__global__ __launch_bounds__(FPS_T) void fps_kernel(const float* __restrict__ x,
                                                    int* __restrict__ idx_out)
{
    __shared__ float sval[32];
    __shared__ int   sidx[32];
    __shared__ int   s_far;

    const int b    = blockIdx.x;
    const int t    = threadIdx.x;
    const int lane = t & 31;
    const int wave = t >> 5;
    const float* xb = x + (size_t)b * N_ * D_;

    // Register-resident points (48 floats) and running distances (8 floats).
    float p[PPT][D_];
    float dist[PPT];
#pragma unroll
    for (int u = 0; u < PPT; ++u) {
        const int pi = t + u * FPS_T;
#pragma unroll
        for (int d = 0; d < D_; ++d) p[u][d] = xb[(size_t)pi * D_ + d];
        dist[u] = 3.0e38f;   // stands in for +inf
    }

    if (t == 0) { idx_out[b * KS] = 0; s_far = 0; }
    __syncthreads();
    int far = 0;

    for (int step = 1; step < KS; ++step) {
        // Broadcast centroid (6 floats, L1-hot scalar-uniform load).
        float c[D_];
#pragma unroll
        for (int d = 0; d < D_; ++d) c[d] = xb[(size_t)far * D_ + d];

        float bestv = -1.0f;
        int   besti = 0;
#pragma unroll
        for (int u = 0; u < PPT; ++u) {
            float dd = 0.0f;
#pragma unroll
            for (int d = 0; d < D_; ++d) {
                const float t0 = p[u][d] - c[d];
                dd = fmaf(t0, t0, dd);
            }
            dist[u] = fminf(dist[u], dd);
            if (dist[u] > bestv) { bestv = dist[u]; besti = t + u * FPS_T; }
        }

        // Wave32 shuffle argmax; ties resolved toward the smaller index
        // (matches jnp.argmax first-max semantics).
#pragma unroll
        for (int off = 16; off > 0; off >>= 1) {
            const float ov = __shfl_down(bestv, off, 32);
            const int   oi = __shfl_down(besti, off, 32);
            if (ov > bestv || (ov == bestv && oi < besti)) { bestv = ov; besti = oi; }
        }
        if (lane == 0) { sval[wave] = bestv; sidx[wave] = besti; }
        __syncthreads();
        if (wave == 0) {
            bestv = sval[lane];
            besti = sidx[lane];
#pragma unroll
            for (int off = 16; off > 0; off >>= 1) {
                const float ov = __shfl_down(bestv, off, 32);
                const int   oi = __shfl_down(besti, off, 32);
                if (ov > bestv || (ov == bestv && oi < besti)) { bestv = ov; besti = oi; }
            }
            if (lane == 0) { s_far = besti; idx_out[b * KS + step] = besti; }
        }
        __syncthreads();
        far = s_far;
    }
}

// ---------------------------------------------------------------------------
// Gather sampled points and emit the layer-1 activation [M_, 32] in f16,
// zero-padded from K=6 to K=32 so the GEMM K-loop is uniform.
// ---------------------------------------------------------------------------
__global__ __launch_bounds__(256) void gather_kernel(const float* __restrict__ x,
                                                     const int* __restrict__ idx,
                                                     half_t* __restrict__ act)
{
    const int gid = blockIdx.x * 256 + threadIdx.x;   // over M_*32 elements
    const int row = gid >> 5;
    const int col = gid & 31;
    const int b   = row / KS;
    const int pi  = idx[row];
    const float v = (col < D_) ? x[((size_t)b * N_ + pi) * D_ + col] : 0.0f;
    act[(size_t)row * 32 + col] = (half_t)v;
}

// ---------------------------------------------------------------------------
// One MLP layer as a WMMA GEMM + sin epilogue.
//   actIn : [M_, KP]  f16 row-major (KP = K padded to multiple of 32)
//   W     : [COUT, CIN] f32 row-major (B = W^T on the fly, f32->f16 cvt)
//   out   : sin(actIn @ W^T + bias)  -> f16 actOut, or f32 d_out if LAST
// 8 waves/block, each wave owns a 16x16 tile; block tile = 128 rows x 16 cols.
// Per-lane A/B K order is contiguous (k = 16*half + e), not the ISA table order; D per ISA (wave32).
// ---------------------------------------------------------------------------
template<int CIN, int KP, int COUT, bool LAST>
__global__ __launch_bounds__(256) void layer_kernel(const half_t* __restrict__ actIn,
                                                    const float*  __restrict__ W,
                                                    const float*  __restrict__ bias,
                                                    half_t* __restrict__ actOut,
                                                    float*  __restrict__ outF32)
{
    const int lane    = threadIdx.x & 31;
    const int wave    = threadIdx.x >> 5;
    const int half_hi = lane >> 4;                  // 0: lanes 0-15, 1: lanes 16-31
    const int row0    = blockIdx.x * 128 + wave * 16;
    const int col0    = blockIdx.y * 16;
    const int mrow    = row0 + (lane & 15);         // A-matrix row for this lane
    const int ncol    = col0 + (lane & 15);         // B-matrix column for this lane

    v8f acc = {};

    for (int kk = 0; kk < KP; kk += 32) {
        // ---- A tile 16x32 f16: lane half selects K 0-15 vs 16-31 ----
        v16h a;
        const half_t* ap = actIn + (size_t)mrow * KP + kk + half_hi * 16;
        union { float4 f; half_t h[8]; } u0, u1;
        u0.f = *(const float4*)(ap);        // 16B aligned: KP mult of 32 halves
        u1.f = *(const float4*)(ap + 8);
#pragma unroll
        for (int e = 0; e < 8; ++e) { a[e] = u0.h[e]; a[8 + e] = u1.h[e]; }

        // ---- B tile 32x16 f16 = W^T slice: 16 consecutive K per lane ----
        v16h bm;
        const int kb = kk + half_hi * 16;
        const float* wp = W + (size_t)ncol * CIN + kb;
#pragma unroll
        for (int e = 0; e < 16; ++e) {
            // compile-time elided guard except for the K=6 first layer
            bm[e] = ((CIN % 32) == 0 || (kb + e) < CIN) ? (half_t)wp[e] : (half_t)0;
        }

        acc = __builtin_amdgcn_wmma_f32_16x16x32_f16(
                  false, a, false, bm, (short)0, acc, false, false);
    }

    // Epilogue: D element (v, lane) = row (v + 8*half_hi), col (lane&15).
    const float bn = bias[ncol];
#pragma unroll
    for (int v = 0; v < 8; ++v) {
        const int m = row0 + v + half_hi * 8;
        const float r = __sinf(acc[v] + bn);       // w0 == 1.0
        if (LAST) outF32[(size_t)m * COUT + ncol] = r;
        else      actOut[(size_t)m * COUT + ncol] = (half_t)r;
    }
}

// ---------------------------------------------------------------------------
extern "C" void kernel_launch(void* const* d_in, const int* in_sizes, int n_in,
                              void* d_out, int out_size, void* d_ws, size_t ws_size,
                              hipStream_t stream)
{
    const float* x  = (const float*)d_in[0];
    const float* W1 = (const float*)d_in[1];  const float* b1 = (const float*)d_in[2];
    const float* W2 = (const float*)d_in[3];  const float* b2 = (const float*)d_in[4];
    const float* W3 = (const float*)d_in[5];  const float* b3 = (const float*)d_in[6];
    const float* W4 = (const float*)d_in[7];  const float* b4 = (const float*)d_in[8];
    const float* W5 = (const float*)d_in[9];  const float* b5 = (const float*)d_in[10];
    const float* W6 = (const float*)d_in[11]; const float* b6 = (const float*)d_in[12];
    float* out = (float*)d_out;               // [16, 1024, 512]

    // Workspace carve-out (~13.1 MB): idx | gathered f16 input | act ping-pong.
    uint8_t* ws = (uint8_t*)d_ws;
    int*    idx  = (int*)ws;                                          // 64 KB
    half_t* aIn  = (half_t*)(ws + (64u << 10));                       // 1 MB (16384x32)
    half_t* act0 = (half_t*)(ws + (64u << 10) + (1u << 20));          // 8 MB (<=16384x256)
    half_t* act1 = (half_t*)(ws + (64u << 10) + (1u << 20) + (8u << 20)); // 4 MB (<=16384x128)

    // 1) FPS: 16 batches, one WGP-filling block each (latency-bound part).
    fps_kernel<<<B_, FPS_T, 0, stream>>>(x, idx);

    // 2) Gather sampled points -> padded f16 layer-1 input.
    gather_kernel<<<(M_ * 32) / 256, 256, 0, stream>>>(x, idx, aIn);

    // 3) Six WMMA GEMM layers with fused sin epilogue (M=16384 everywhere).
    layer_kernel<  6,  32,  32, false><<<dim3(M_/128,  2), 256, 0, stream>>>(aIn,  W1, b1, act0, nullptr);
    layer_kernel< 32,  32,  64, false><<<dim3(M_/128,  4), 256, 0, stream>>>(act0, W2, b2, act1, nullptr);
    layer_kernel< 64,  64,  64, false><<<dim3(M_/128,  4), 256, 0, stream>>>(act1, W3, b3, act0, nullptr);
    layer_kernel< 64,  64, 128, false><<<dim3(M_/128,  8), 256, 0, stream>>>(act0, W4, b4, act1, nullptr);
    layer_kernel<128, 128, 256, false><<<dim3(M_/128, 16), 256, 0, stream>>>(act1, W5, b5, act0, nullptr);
    layer_kernel<256, 256, 512, true ><<<dim3(M_/128, 32), 256, 0, stream>>>(act0, W6, b6, nullptr, out);
}
